// RnnModel_78383153152376
// MI455X (gfx1250) — matrix operand order repaired and checked by reading, not compiled, not run on hardware
//
#include <hip/hip_runtime.h>

#define VOCAB  32000
#define EDIM   256
#define HLDIM  1024
#define HCDIM  256
#define BATCH  512
#define NSTEPS 100
#define XDIM   (EDIM + HCDIM)   // 512
#define G3     (3 * HLDIM)      // 3072

typedef __attribute__((ext_vector_type(16))) __bf16 v16bf;
typedef __attribute__((ext_vector_type(8)))  float  v8f;
typedef int v4i_ __attribute__((__vector_size__(16)));

union FragAB { unsigned int u[8]; v16bf v; };
union FragC  { float f[8]; v8f v; };

// ---------- small device helpers ----------

__device__ __forceinline__ unsigned short f2bf(float x) {
  unsigned int u = __float_as_uint(x);
  unsigned int r = u + 0x7FFFu + ((u >> 16) & 1u);   // round-to-nearest-even
  return (unsigned short)(r >> 16);
}

__device__ __forceinline__ float sigmoidf(float x) {
  return 1.0f / (1.0f + __expf(-x));
}

__device__ __forceinline__ float gumbel_noise(unsigned t, unsigned b, unsigned v) {
  unsigned s = t * 0x9E3779B9u ^ (b * 0x85EBCA6Bu) ^ (v * 0xC2B2AE35u);
  s ^= s >> 16; s *= 0x7FEB352Du;
  s ^= s >> 15; s *= 0x846CA68Bu;
  s ^= s >> 16;
  float u = (float)(s >> 8) * (1.0f / 16777216.0f) + 1e-12f;
  return -__logf(-__logf(u));
}

__device__ __forceinline__ unsigned long long packkey(float f, int idx) {
  unsigned u = __float_as_uint(f);
  u = (u & 0x80000000u) ? ~u : (u | 0x80000000u);
  return ((unsigned long long)u << 32) | (unsigned)idx;
}

// ---------- CDNA5 async global->LDS copy (ASYNCcnt-tracked) ----------

__device__ __forceinline__ void async_g2l_b128(const void* g, void* l) {
#if defined(__has_builtin) && __has_builtin(__builtin_amdgcn_global_load_async_to_lds_b128)
  __builtin_amdgcn_global_load_async_to_lds_b128(
      (__attribute__((address_space(1))) v4i_*)(v4i_*)g,
      (__attribute__((address_space(3))) v4i_*)(v4i_*)l, 0, 0);
#else
  unsigned loff = (unsigned)(unsigned long long)
      (__attribute__((address_space(3))) char*)(char*)l;
  asm volatile("global_load_async_to_lds_b128 %0, %1, off"
               :: "v"(loff), "v"(g) : "memory");
#endif
}

__device__ __forceinline__ void wait_async0() {
#if defined(__has_builtin) && __has_builtin(__builtin_amdgcn_s_wait_asynccnt)
  __builtin_amdgcn_s_wait_asynccnt(0);
#else
  asm volatile("s_wait_asynccnt 0x0" ::: "memory");
#endif
}
__device__ __forceinline__ void wait_async1() {
#if defined(__has_builtin) && __has_builtin(__builtin_amdgcn_s_wait_asynccnt)
  __builtin_amdgcn_s_wait_asynccnt(1);
#else
  asm volatile("s_wait_asynccnt 0x1" ::: "memory");
#endif
}
__device__ __forceinline__ void wait_async2() {
#if defined(__has_builtin) && __has_builtin(__builtin_amdgcn_s_wait_asynccnt)
  __builtin_amdgcn_s_wait_asynccnt(2);
#else
  asm volatile("s_wait_asynccnt 0x2" ::: "memory");
#endif
}

// ---------- WMMA fragment loads (layouts per CDNA5 ISA 7.12.2) ----------

// A 16x32 bf16, row-major, leading dim ld.
__device__ __forceinline__ v16bf load_a16(const unsigned short* __restrict__ p, int ld) {
  int lane = threadIdx.x & 31;
  int row  = lane & 15;
  int kh   = (lane >> 4) << 3;
  const unsigned short* rp = p + row * ld + kh;
  FragAB f;
#pragma unroll
  for (int r = 0; r < 8; ++r) {
    int k = (r < 4) ? (2 * r) : (2 * r + 8);
    f.u[r] = *(const unsigned int*)(rp + k);
  }
  return f.v;
}

// B 32x16 bf16 from row-major W [N x K]: B[k][n] = W[n0+n][k0+k].
// Works on global or LDS-backed pointers (inlined; addrspace inferred -> ds_load).
__device__ __forceinline__ v16bf load_b16(const unsigned short* __restrict__ p, int ld) {
  int lane = threadIdx.x & 31;
  int nn   = lane & 15;
  int kh   = (lane >> 4) << 3;
  const unsigned short* rp = p + nn * ld + kh;
  FragAB f;
#pragma unroll
  for (int r = 0; r < 8; ++r)
    f.u[r] = *(const unsigned int*)(rp + ((r < 4) ? (2 * r) : (2 * r + 8)));
  return f.v;
}

__device__ __forceinline__ v8f wmma_bf16(v16bf a, v16bf b, v8f c) {
  return __builtin_amdgcn_wmma_f32_16x16x32_bf16(
      false, a, false, b, (short)0, c, false, false);
}

// ---------- one-time setup kernels ----------

__global__ void convert_f32_bf16(const float* __restrict__ src,
                                 unsigned short* __restrict__ dst, int n) {
  int i = blockIdx.x * blockDim.x + threadIdx.x;
  int stride = gridDim.x * blockDim.x;
  for (; i < n; i += stride) dst[i] = f2bf(src[i]);
}

__global__ void init_h(const float* __restrict__ h0, float* __restrict__ hf,
                       unsigned short* __restrict__ hbf, float* __restrict__ out_hs) {
  int i = blockIdx.x * blockDim.x + threadIdx.x;
  if (i < BATCH * HLDIM) {
    float v = h0[i];
    hf[i] = v;
    hbf[i] = f2bf(v);
    out_hs[i] = v;
  }
}

__global__ void init_inpt(const int* __restrict__ inpt0, int* __restrict__ buf) {
  int b = threadIdx.x;
  if (b < BATCH) buf[b] = inpt0[b];
}

// ---------- per-step kernels ----------

__global__ void gather_x(const float* __restrict__ emb, const float* __restrict__ ctx,
                         const int* __restrict__ inpt, unsigned short* __restrict__ x,
                         unsigned long long* __restrict__ keys) {
  int b = blockIdx.x;
  int j = threadIdx.x;
  int w = inpt[b];
  x[b * XDIM + j]        = f2bf(emb[w * EDIM + j]);
  x[b * XDIM + EDIM + j] = f2bf(ctx[b * HCDIM + j]);
  if (j == 0) keys[b] = 0ull;
}

// GRU: block = 8 waves sharing one HL column tile (nt), 8 batch row tiles.
// Gate B-tiles (3 x 1KB per K-step) double-buffered in LDS via async copy.
// Every wave issues exactly ONE async b128 per stage (lanes 0..23 active),
// so ASYNCcnt is wave-uniform and waits need no branching.
__global__ void __launch_bounds__(256)
gru_step(const unsigned short* __restrict__ x,       // [B, XDIM]
         const unsigned short* __restrict__ wih,     // [G3, XDIM]
         const unsigned short* __restrict__ whh,     // [G3, HLDIM]
         const unsigned short* __restrict__ hin_bf,  // [B, HLDIM]
         const float* __restrict__ hin_f,
         const float* __restrict__ bih, const float* __restrict__ bhh,
         float* __restrict__ hout_f, unsigned short* __restrict__ hout_bf,
         float* __restrict__ out_hs_slot) {
  __shared__ __align__(16) unsigned short bsm[2][3 * 16 * 32];  // 2 x 3KB

  int wid  = threadIdx.x >> 5;
  int lane = threadIdx.x & 31;
  int nt = blockIdx.x & 63;          // HL tile (shared by block)
  int mg = blockIdx.x >> 6;          // 0..3
  int mt = mg * 8 + wid;             // per-wave batch tile

  auto stage3 = [&](const unsigned short* w, int ld, int kt, int buf) {
    if (lane < 24) {                     // 8 waves x 24 lanes = 192 x 16B = 3KB
      int idx  = wid * 24 + lane;        // chunk id
      int g    = idx >> 6;               // gate 0..2
      int c    = idx & 63;               // chunk within 16x32 tile
      int row  = c >> 2;
      int coff = (c & 3) * 8;            // ushort offset in row
      const unsigned short* src =
          w + (size_t)(g * HLDIM + nt * 16 + row) * ld + kt * 32 + coff;
      async_g2l_b128(src, &bsm[buf][g * 512 + row * 32 + coff]);
    }
  };

  v8f air = {}, aiz = {}, ain = {}, ahr = {}, ahz = {}, ahn = {};
  const unsigned short* xbase = x + mt * 16 * XDIM;
  const unsigned short* hbase = hin_bf + mt * 16 * HLDIM;

  const int NX = XDIM / 32;    // 16
  const int NH = HLDIM / 32;   // 32

  stage3(wih, XDIM, 0, 0);
  // ---- x-phase: gi = x @ w_ih^T ----
  for (int s = 0; s < NX; ++s) {
    if (s + 1 < NX) stage3(wih, XDIM, s + 1, (s + 1) & 1);
    else            stage3(whh, HLDIM, 0, 0);   // NX even -> h kt=0 lands in buf 0
    wait_async1();
    __syncthreads();
    const unsigned short* bb = &bsm[s & 1][0];
    v16bf a = load_a16(xbase + s * 32, XDIM);
    air = wmma_bf16(a, load_b16(bb + 0,    32), air);
    aiz = wmma_bf16(a, load_b16(bb + 512,  32), aiz);
    ain = wmma_bf16(a, load_b16(bb + 1024, 32), ain);
    __syncthreads();
  }
  // ---- h-phase: gh = h @ w_hh^T ----
  for (int kt = 0; kt < NH; ++kt) {
    if (kt + 1 < NH) { stage3(whh, HLDIM, kt + 1, (kt + 1) & 1); wait_async1(); }
    else             { wait_async0(); }
    __syncthreads();
    const unsigned short* bb = &bsm[kt & 1][0];
    v16bf a = load_a16(hbase + kt * 32, HLDIM);
    ahr = wmma_bf16(a, load_b16(bb + 0,    32), ahr);
    ahz = wmma_bf16(a, load_b16(bb + 512,  32), ahz);
    ahn = wmma_bf16(a, load_b16(bb + 1024, 32), ahn);
    __syncthreads();
  }

  int n  = lane & 15;
  int mh = (lane >> 4) << 3;
  int col = nt * 16 + n;
  float bir = bih[col], biz = bih[HLDIM + col], bin_ = bih[2 * HLDIM + col];
  float bhr = bhh[col], bhz = bhh[HLDIM + col], bhn  = bhh[2 * HLDIM + col];

  FragC fir, fiz, fin, fhr, fhz, fhn;
  fir.v = air; fiz.v = aiz; fin.v = ain; fhr.v = ahr; fhz.v = ahz; fhn.v = ahn;
#pragma unroll
  for (int r = 0; r < 8; ++r) {
    int m = mt * 16 + mh + r;
    float rg = sigmoidf(fir.f[r] + bir + fhr.f[r] + bhr);
    float zg = sigmoidf(fiz.f[r] + biz + fhz.f[r] + bhz);
    float ng = tanhf(fin.f[r] + bin_ + rg * (fhn.f[r] + bhn));
    float ho = hin_f[m * HLDIM + col];
    float hn2 = (1.0f - zg) * ng + zg * ho;
    hout_f[m * HLDIM + col] = hn2;
    hout_bf[m * HLDIM + col] = f2bf(hn2);
    out_hs_slot[m * HLDIM + col] = hn2;
  }
}

// dec = bf16((h @ dec_W^T + dec_b) / temp)
__global__ void __launch_bounds__(256)
dec_step(const unsigned short* __restrict__ h_bf, const unsigned short* __restrict__ decw,
         const float* __restrict__ decb, const int* __restrict__ temp,
         unsigned short* __restrict__ dec_bf) {
  int wave = blockIdx.x * (blockDim.x >> 5) + (threadIdx.x >> 5);
  int mt = wave >> 4;
  int nt = wave & 15;
  v8f acc = {};
  const unsigned short* abase = h_bf + mt * 16 * HLDIM;
  const unsigned short* bbase = decw + nt * 16 * HLDIM;
  for (int kt = 0; kt < HLDIM / 32; ++kt) {
    __builtin_prefetch(bbase + (kt + 1) * 32, 0, 1);
    acc = wmma_bf16(load_a16(abase + kt * 32, HLDIM),
                    load_b16(bbase + kt * 32, HLDIM), acc);
  }
  int lane = threadIdx.x & 31;
  int n = lane & 15, mh = (lane >> 4) << 3;
  int col = nt * 16 + n;
  float invT = 1.0f / (float)temp[0];
  float bb = decb[col];
  FragC c; c.v = acc;
#pragma unroll
  for (int r = 0; r < 8; ++r) {
    int m = mt * 16 + mh + r;
    dec_bf[m * EDIM + col] = f2bf((c.f[r] + bb) * invT);
  }
}

// logits GEMM fused with Gumbel-argmax; emb vocab tiles (8KB) staged in LDS,
// shared by 8 waves covering 8 batch row tiles. Double-buffered async copy.
#define NT_PER_CHUNK 40
#define NCHUNKS      (VOCAB / 16 / NT_PER_CHUNK)   // 50
__global__ void __launch_bounds__(256)
logits_argmax(const unsigned short* __restrict__ dec_bf,
              const unsigned short* __restrict__ emb_bf,
              unsigned long long* __restrict__ keys, int step) {
  __shared__ __align__(16) unsigned short esm[2][16 * 256];   // 2 x 8KB

  int wid  = threadIdx.x >> 5;
  int lane = threadIdx.x & 31;
  int chunk = blockIdx.x % NCHUNKS;
  int rg    = blockIdx.x / NCHUNKS;   // 0..3
  int mt    = rg * 8 + wid;

  // every thread deposits 2 x 16B of the contiguous 8KB vocab tile
  auto stage = [&](int i, int buf) {
    int nt = chunk * NT_PER_CHUNK + i;
    const unsigned short* src = emb_bf + (size_t)nt * 16 * EDIM;
    int o0 = threadIdx.x * 8;          // ushort offset of 16B chunk
    async_g2l_b128(src + o0,        &esm[buf][o0]);
    async_g2l_b128(src + o0 + 2048, &esm[buf][o0 + 2048]);
  };

  // hoist all 8 dec A-fragments: reused for the whole vocab chunk
  v16bf afr[8];
  const unsigned short* abase = dec_bf + mt * 16 * EDIM;
#pragma unroll
  for (int kt = 0; kt < 8; ++kt) afr[kt] = load_a16(abase + kt * 32, EDIM);

  int n = lane & 15, mh = (lane >> 4) << 3;
  float bestk[8]; int besti[8];
#pragma unroll
  for (int r = 0; r < 8; ++r) { bestk[r] = -3.4e38f; besti[r] = 0; }

  stage(0, 0);
  for (int i = 0; i < NT_PER_CHUNK; ++i) {
    if (i + 1 < NT_PER_CHUNK) { stage(i + 1, (i + 1) & 1); wait_async2(); }
    else                      { wait_async0(); }
    __syncthreads();
    const unsigned short* bb = &esm[i & 1][0];
    v8f acc = {};
#pragma unroll
    for (int kt = 0; kt < 8; ++kt)
      acc = wmma_bf16(afr[kt], load_b16(bb + kt * 32, EDIM), acc);
    FragC c; c.v = acc;
    int v0 = (chunk * NT_PER_CHUNK + i) * 16 + n;
#pragma unroll
    for (int r = 0; r < 8; ++r) {
      int m = mt * 16 + mh + r;
      float key = c.f[r] + gumbel_noise((unsigned)step, (unsigned)m, (unsigned)v0);
      if (key > bestk[r]) { bestk[r] = key; besti[r] = v0; }
    }
    __syncthreads();
  }

#pragma unroll
  for (int r = 0; r < 8; ++r) {
    float bk = bestk[r]; int bi = besti[r];
#pragma unroll
    for (int mask = 8; mask >= 1; mask >>= 1) {
      float ok = __shfl_xor(bk, mask, 32);
      int   oi = __shfl_xor(bi, mask, 32);
      if (ok > bk) { bk = ok; bi = oi; }
    }
    if ((lane & 15) == 0) {
      int m = mt * 16 + mh + r;
      atomicMax(&keys[m], packkey(bk, bi));
    }
  }
}

__global__ void sample_step(const unsigned long long* __restrict__ keys,
                            float* __restrict__ outs, int* __restrict__ inpt_buf,
                            int step) {
  int b = threadIdx.x;
  if (b < BATCH) {
    int idx = (int)(unsigned)(keys[b] & 0xFFFFFFFFull);
    outs[step * BATCH + b] = (float)idx;
    inpt_buf[b] = idx;
  }
}

// ---------- host orchestration ----------

extern "C" void kernel_launch(void* const* d_in, const int* in_sizes, int n_in,
                              void* d_out, int out_size, void* d_ws, size_t ws_size,
                              hipStream_t stream) {
  (void)in_sizes; (void)n_in; (void)out_size; (void)ws_size;
  const float* lang_h0  = (const float*)d_in[0];
  const float* ctx_h    = (const float*)d_in[1];
  const float* word_emb = (const float*)d_in[2];
  const float* dec_W    = (const float*)d_in[3];
  const float* dec_b    = (const float*)d_in[4];
  const float* w_ih     = (const float*)d_in[5];
  const float* w_hh     = (const float*)d_in[6];
  const float* b_ih     = (const float*)d_in[7];
  const float* b_hh     = (const float*)d_in[8];
  const int*   inpt0    = (const int*)d_in[9];
  const int*   temp     = (const int*)d_in[10];

  char* ws = (char*)d_ws;
  size_t off = 0;
  auto alloc = [&](size_t bytes) -> void* {
    void* p = ws + off;
    off += (bytes + 255) & ~(size_t)255;
    return p;
  };

  unsigned short* wih_bf  = (unsigned short*)alloc((size_t)G3 * XDIM * 2);
  unsigned short* whh_bf  = (unsigned short*)alloc((size_t)G3 * HLDIM * 2);
  unsigned short* decw_bf = (unsigned short*)alloc((size_t)EDIM * HLDIM * 2);
  unsigned short* emb_bf  = (unsigned short*)alloc((size_t)VOCAB * EDIM * 2);
  unsigned short* x_bf    = (unsigned short*)alloc((size_t)BATCH * XDIM * 2);
  float*          hf0     = (float*)alloc((size_t)BATCH * HLDIM * 4);
  float*          hf1     = (float*)alloc((size_t)BATCH * HLDIM * 4);
  unsigned short* hbf0    = (unsigned short*)alloc((size_t)BATCH * HLDIM * 2);
  unsigned short* hbf1    = (unsigned short*)alloc((size_t)BATCH * HLDIM * 2);
  unsigned short* dec_bf  = (unsigned short*)alloc((size_t)BATCH * EDIM * 2);
  unsigned long long* keys = (unsigned long long*)alloc((size_t)BATCH * 8);
  int*            inpt_buf = (int*)alloc((size_t)BATCH * 4);

  float* hf[2]  = {hf0, hf1};
  unsigned short* hbf[2] = {hbf0, hbf1};

  float* outs = (float*)d_out;
  float* hs   = (float*)d_out + NSTEPS * BATCH;

  {
    int n;
    n = G3 * XDIM;    convert_f32_bf16<<<(n + 1023) / 1024, 256, 0, stream>>>(w_ih, wih_bf, n);
    n = G3 * HLDIM;   convert_f32_bf16<<<(n + 1023) / 1024, 256, 0, stream>>>(w_hh, whh_bf, n);
    n = EDIM * HLDIM; convert_f32_bf16<<<(n + 1023) / 1024, 256, 0, stream>>>(dec_W, decw_bf, n);
    n = VOCAB * EDIM; convert_f32_bf16<<<(n + 1023) / 1024, 256, 0, stream>>>(word_emb, emb_bf, n);
    init_h<<<(BATCH * HLDIM + 255) / 256, 256, 0, stream>>>(lang_h0, hf[0], hbf[0], hs);
    init_inpt<<<1, BATCH, 0, stream>>>(inpt0, inpt_buf);
  }

  for (int t = 0; t <= NSTEPS; ++t) {
    int p = t & 1;
    gather_x<<<BATCH, EDIM, 0, stream>>>(word_emb, ctx_h, inpt_buf, x_bf, keys);
    gru_step<<<(32 * 64) / 8, 256, 0, stream>>>(
        x_bf, wih_bf, whh_bf, hbf[p], hf[p], b_ih, b_hh,
        hf[1 - p], hbf[1 - p], hs + (size_t)(t + 1) * BATCH * HLDIM);
    if (t < NSTEPS) {
      dec_step<<<(32 * 16) / 8, 256, 0, stream>>>(hbf[1 - p], decw_bf, dec_b, temp, dec_bf);
      logits_argmax<<<4 * NCHUNKS, 256, 0, stream>>>(dec_bf, emb_bf, keys, t);
      sample_step<<<1, BATCH, 0, stream>>>(keys, outs, inpt_buf, t);
    }
  }
}
